// CT_StochasticGraphDS_55740085567596
// MI455X (gfx1250) — matrix operand order repaired and checked by reading, not compiled, not run on hardware
//
#include <hip/hip_runtime.h>
#include <hip/hip_bf16.h>
#include <math.h>

// ---------------- Problem constants ----------------
#define NN   4096   // nodes
#define DD   8      // state dim
#define HH   64     // MLP hidden
#define INP  16     // input dim
#define RDD  32     // readout dim
#define BB   8      // batch
#define SS   16     // time steps
#define KK   4      // Euler substeps
#define BND  (BB*NN*DD)        // 262144 floats per [B,N,D] slab
#define GC   (BB*DD)           // 64 GEMM columns

typedef __bf16 bf16_t;
typedef __attribute__((ext_vector_type(8)))  __bf16 v8bf;
typedef __attribute__((ext_vector_type(16))) __bf16 v16bf;
typedef __attribute__((ext_vector_type(8)))  float  v8f;

static __device__ __forceinline__ v16bf cat8(v8bf lo, v8bf hi) {
  return __builtin_shufflevector(lo, hi, 0,1,2,3,4,5,6,7,8,9,10,11,12,13,14,15);
}

static __device__ __forceinline__ float sigmoidf_(float x) {
  return 1.0f / (1.0f + __expf(-x));
}

// ---------------- A f32 -> bf16 (once per launch) ----------------
__global__ __launch_bounds__(256) void convA_kernel(const float* __restrict__ A,
                                                    bf16_t* __restrict__ Abf) {
  size_t i = ((size_t)blockIdx.x * 256 + threadIdx.x) * 8;   // 8192 blocks cover 16.7M
  float4 f0 = *(const float4*)(A + i);
  float4 f1 = *(const float4*)(A + i + 4);
  v8bf o;
  o[0] = (bf16_t)f0.x; o[1] = (bf16_t)f0.y; o[2] = (bf16_t)f0.z; o[3] = (bf16_t)f0.w;
  o[4] = (bf16_t)f1.x; o[5] = (bf16_t)f1.y; o[6] = (bf16_t)f1.z; o[7] = (bf16_t)f1.w;
  *(v8bf*)(Abf + i) = o;
}

// ---------------- Graph GEMM: G[4096][64] = Abf[4096][4096] @ Ht^T ----------------
// Ht is the B operand stored transposed: Ht[c][j], c = b*8+d (64 rows), j = node (4096).
// Each wave: 16 rows x 32 cols of G, K-loop in chunks of 32 with v_wmma_f32_16x16x32_bf16.
__global__ __launch_bounds__(256) void gemm_graph_bf16(const bf16_t* __restrict__ Abf,
                                                       const bf16_t* __restrict__ Ht,
                                                       float* __restrict__ G) {
  const int lane = threadIdx.x & 31;
  const int wv   = threadIdx.x >> 5;
  const int wg   = blockIdx.x * 8 + wv;       // 0..511 (64 blocks x 8 waves)
  const int rowg = wg >> 1;                   // 0..255 -> 16-row group
  const int c0   = (wg & 1) * 32;             // 0 or 32
  const int r    = lane & 15;
  const int kh   = lane >> 4;

  // A-frag addressing (contiguous k = 16*kh + s, matching B): lane holds row rowg*16+r,
  // K = kh*16..kh*16+15 within each 32-chunk.
  const bf16_t* Ap  = Abf + (size_t)(rowg * 16 + r) * NN + kh * 16;
  // B-frag addressing (contiguous k = 16*kh + s, matching A): lane holds col c, K = kh*16..kh*16+15.
  const bf16_t* Bp0 = Ht + (size_t)(c0      + r) * NN + kh * 16;
  const bf16_t* Bp1 = Ht + (size_t)(c0 + 16 + r) * NN + kh * 16;

  v8f acc0 = {}; v8f acc1 = {};
  #pragma unroll 2
  for (int k = 0; k < NN; k += 32) {
    __builtin_prefetch(Ap + k + 2048, 0, 1);   // global_prefetch_b8 on the A stream
    v16bf a  = cat8(*(const v8bf*)(Ap  + k), *(const v8bf*)(Ap  + k + 8));
    v16bf b0 = cat8(*(const v8bf*)(Bp0 + k), *(const v8bf*)(Bp0 + k + 8));
    v16bf b1 = cat8(*(const v8bf*)(Bp1 + k), *(const v8bf*)(Bp1 + k + 8));
    acc0 = __builtin_amdgcn_wmma_f32_16x16x32_bf16(false, a, false, b0,
                                                   (short)0, acc0, false, false);
    acc1 = __builtin_amdgcn_wmma_f32_16x16x32_bf16(false, a, false, b1,
                                                   (short)0, acc1, false, false);
  }
  // C/D layout: VGPR v -> row (kh*8 + v), col = lane&15.
  const int n0 = lane & 15;
  const int m0 = rowg * 16 + kh * 8;
  #pragma unroll
  for (int v = 0; v < 8; ++v) {
    G[(size_t)(m0 + v) * GC + c0 + n0]      = acc0[v];
    G[(size_t)(m0 + v) * GC + c0 + 16 + n0] = acc1[v];
  }
}

// ---------------- Perturbation: hp = h + relu([x,h]@pW1+pb1)@pW2+pb2 ----------------
__global__ __launch_bounds__(256) void perturb_kernel(
    const float* __restrict__ hin, const float* __restrict__ x,
    const float* __restrict__ pW1, const float* __restrict__ pb1,
    const float* __restrict__ pW2, const float* __restrict__ pb2,
    float* __restrict__ hout, bf16_t* __restrict__ Ht) {
  __shared__ float sW1[(INP + DD) * HH];   // 24x64
  __shared__ float sW2[HH * DD];           // 64x8
  __shared__ float sb1[HH];
  __shared__ float sb2[DD];
  for (int i = threadIdx.x; i < (INP + DD) * HH; i += 256) sW1[i] = pW1[i];
  for (int i = threadIdx.x; i < HH * DD; i += 256)         sW2[i] = pW2[i];
  if (threadIdx.x < HH) sb1[threadIdx.x] = pb1[threadIdx.x];
  if (threadIdx.x < DD) sb2[threadIdx.x] = pb2[threadIdx.x];
  __syncthreads();

  const int tid = blockIdx.x * 256 + threadIdx.x;  // 0..32767 == b*N+n
  const int b = tid >> 12, n = tid & (NN - 1);

  float comb[INP + DD];
  #pragma unroll
  for (int m = 0; m < INP; ++m) comb[m] = x[(size_t)tid * INP + m];
  float h[DD];
  #pragma unroll
  for (int d = 0; d < DD; ++d) { h[d] = hin[(size_t)tid * DD + d]; comb[INP + d] = h[d]; }

  float delta[DD];
  #pragma unroll
  for (int d = 0; d < DD; ++d) delta[d] = sb2[d];
  for (int j = 0; j < HH; ++j) {
    float u = sb1[j];
    #pragma unroll
    for (int m = 0; m < INP + DD; ++m) u += comb[m] * sW1[m * HH + j];
    float a = fmaxf(u, 0.0f);
    #pragma unroll
    for (int d = 0; d < DD; ++d) delta[d] += a * sW2[j * DD + d];
  }
  #pragma unroll
  for (int d = 0; d < DD; ++d) {
    float hp = h[d] + delta[d];
    hout[(size_t)tid * DD + d] = hp;
    Ht[(size_t)(b * DD + d) * NN + n] = (bf16_t)hp;   // transposed bf16 B operand
  }
}

// ---------------- Euler substep: h += drift(t, A@h)*dt + diff(t, h)*(sqdt*z) ----------------
__global__ __launch_bounds__(256) void euler_kernel(
    const float* __restrict__ hin, const float* __restrict__ G,
    const float* __restrict__ z, const float* __restrict__ times, int s, int k,
    const float* __restrict__ dW1, const float* __restrict__ db1,
    const float* __restrict__ dW2, const float* __restrict__ db2,
    const float* __restrict__ fW1, const float* __restrict__ fb1,
    const float* __restrict__ fW2, const float* __restrict__ fb2,
    float* __restrict__ hout, bf16_t* __restrict__ Ht) {
  __shared__ float sdW1[(DD + 1) * HH], sdW2[HH * DD], sdb1[HH], sdb2[DD];
  __shared__ float sfW1[(DD + 1) * HH], sfW2[HH * DD], sfb1[HH], sfb2[DD];
  for (int i = threadIdx.x; i < (DD + 1) * HH; i += 256) { sdW1[i] = dW1[i]; sfW1[i] = fW1[i]; }
  for (int i = threadIdx.x; i < HH * DD; i += 256)       { sdW2[i] = dW2[i]; sfW2[i] = fW2[i]; }
  if (threadIdx.x < HH) { sdb1[threadIdx.x] = db1[threadIdx.x]; sfb1[threadIdx.x] = fb1[threadIdx.x]; }
  if (threadIdx.x < DD) { sdb2[threadIdx.x] = db2[threadIdx.x]; sfb2[threadIdx.x] = fb2[threadIdx.x]; }
  __syncthreads();

  const int tid = blockIdx.x * 256 + threadIdx.x;  // 0..32767 == b*N+n
  const int b = tid >> 12, n = tid & (NN - 1);

  const float t0 = times[s], t1 = times[s + 1];
  const float dt = (t1 - t0) * (1.0f / (float)KK);
  const float t  = t0 + (float)k * dt;
  const float sq = sqrtf(dt);

  float h[DD], g[DD];
  #pragma unroll
  for (int d = 0; d < DD; ++d) h[d] = hin[(size_t)tid * DD + d];
  #pragma unroll
  for (int d = 0; d < DD; ++d) g[d] = G[(size_t)n * GC + b * DD + d];

  float dr[DD], df[DD];
  #pragma unroll
  for (int d = 0; d < DD; ++d) { dr[d] = sdb2[d]; df[d] = sfb2[d]; }

  for (int j = 0; j < HH; ++j) {
    float u = sdb1[j];
    float w = sfb1[j];
    #pragma unroll
    for (int m = 0; m < DD; ++m) { u += g[m] * sdW1[m * HH + j]; w += h[m] * sfW1[m * HH + j]; }
    u += t * sdW1[DD * HH + j];
    w += t * sfW1[DD * HH + j];
    float au = tanhf(u);
    float aw = sigmoidf_(w);
    #pragma unroll
    for (int d = 0; d < DD; ++d) { dr[d] += au * sdW2[j * DD + d]; df[d] += aw * sfW2[j * DD + d]; }
  }

  const float* zp = z + (size_t)tid * DD;
  #pragma unroll
  for (int d = 0; d < DD; ++d) {
    float dif = 0.1f * sigmoidf_(df[d]);
    float hn  = h[d] + dr[d] * dt + dif * (sq * zp[d]);
    hout[(size_t)tid * DD + d] = hn;
    Ht[(size_t)(b * DD + d) * NN + n] = (bf16_t)hn;
  }
}

// ---------------- Readout: relu(traj@rW1+rb1)@rW2+rb2 over all S*B*N rows ----------------
__global__ __launch_bounds__(256) void readout_kernel(
    const float* __restrict__ traj,
    const float* __restrict__ rW1, const float* __restrict__ rb1,
    const float* __restrict__ rW2, const float* __restrict__ rb2,
    float* __restrict__ out) {
  __shared__ float sW1[DD * HH], sW2[HH * RDD], sb1[HH], sb2[RDD];
  for (int i = threadIdx.x; i < DD * HH; i += 256)  sW1[i] = rW1[i];
  for (int i = threadIdx.x; i < HH * RDD; i += 256) sW2[i] = rW2[i];
  if (threadIdx.x < HH)  sb1[threadIdx.x] = rb1[threadIdx.x];
  if (threadIdx.x < RDD) sb2[threadIdx.x] = rb2[threadIdx.x];
  __syncthreads();

  const int tid = blockIdx.x * 256 + threadIdx.x;  // 0..524287
  float h[DD];
  #pragma unroll
  for (int d = 0; d < DD; ++d) h[d] = traj[(size_t)tid * DD + d];
  float o[RDD];
  #pragma unroll
  for (int r = 0; r < RDD; ++r) o[r] = sb2[r];
  for (int j = 0; j < HH; ++j) {
    float u = sb1[j];
    #pragma unroll
    for (int m = 0; m < DD; ++m) u += h[m] * sW1[m * HH + j];
    float a = fmaxf(u, 0.0f);
    #pragma unroll
    for (int r = 0; r < RDD; ++r) o[r] += a * sW2[j * RDD + r];
  }
  #pragma unroll
  for (int r = 0; r < RDD; ++r) out[(size_t)tid * RDD + r] = o[r];
}

// ---------------- Host-side launcher ----------------
extern "C" void kernel_launch(void* const* d_in, const int* in_sizes, int n_in,
                              void* d_out, int out_size, void* d_ws, size_t ws_size,
                              hipStream_t stream) {
  const float* times  = (const float*)d_in[0];
  const float* inputs = (const float*)d_in[1];
  const float* h0     = (const float*)d_in[2];
  const float* A      = (const float*)d_in[3];
  const float* z      = (const float*)d_in[4];
  const float* dW1 = (const float*)d_in[5];
  const float* db1 = (const float*)d_in[6];
  const float* dW2 = (const float*)d_in[7];
  const float* db2 = (const float*)d_in[8];
  const float* fW1 = (const float*)d_in[9];
  const float* fb1 = (const float*)d_in[10];
  const float* fW2 = (const float*)d_in[11];
  const float* fb2 = (const float*)d_in[12];
  const float* pW1 = (const float*)d_in[13];
  const float* pb1 = (const float*)d_in[14];
  const float* pW2 = (const float*)d_in[15];
  const float* pb2 = (const float*)d_in[16];
  const float* rW1 = (const float*)d_in[17];
  const float* rb1 = (const float*)d_in[18];
  const float* rW2 = (const float*)d_in[19];
  const float* rb2 = (const float*)d_in[20];

  float* traj     = (float*)d_out;                        // [S,B,N,D]
  float* readouts = traj + (size_t)SS * BND;              // [S,B,N,RD]

  char* ws = (char*)d_ws;
  bf16_t* Abf  = (bf16_t*)(ws);                           // 32 MiB
  bf16_t* Ht   = (bf16_t*)(ws + (size_t)NN * NN * 2);     // 512 KiB  [64][4096] bf16
  float*  G    = (float*)(ws + (size_t)NN * NN * 2 + (size_t)GC * NN * 2);     // 1 MiB
  float*  hcur = (float*)(ws + (size_t)NN * NN * 2 + (size_t)GC * NN * 2 + (size_t)NN * GC * 4);

  // A -> bf16 once per launch (deterministic; ws is scratch every call)
  convA_kernel<<<(NN * NN) / (256 * 8), 256, 0, stream>>>(A, Abf);
  // traj[0] = h0
  hipMemcpyAsync(traj, h0, (size_t)BND * sizeof(float), hipMemcpyDeviceToDevice, stream);

  for (int s = 0; s < SS - 1; ++s) {
    const float* hprev = traj + (size_t)s * BND;
    perturb_kernel<<<(BB * NN) / 256, 256, 0, stream>>>(
        hprev, inputs + (size_t)s * BB * NN * INP,
        pW1, pb1, pW2, pb2, hcur, Ht);
    for (int k = 0; k < KK; ++k) {
      gemm_graph_bf16<<<64, 256, 0, stream>>>(Abf, Ht, G);
      float* hout = (k == KK - 1) ? (traj + (size_t)(s + 1) * BND) : hcur;
      euler_kernel<<<(BB * NN) / 256, 256, 0, stream>>>(
          hcur, G, z + (size_t)(s * KK + k) * BND, times, s, k,
          dW1, db1, dW2, db2, fW1, fb1, fW2, fb2, hout, Ht);
      // NOTE: on the last substep hout==traj[s+1]; hcur stays the k=KK-2 state,
      // which is never read again (next perturb reads traj[s+1]).
    }
  }
  readout_kernel<<<(SS * BB * NN) / 256, 256, 0, stream>>>(traj, rW1, rb1, rW2, rb2, readouts);
}
